// SelfAttentionUnit_23227183137084
// MI455X (gfx1250) — matrix operand order repaired and checked by reading, not compiled, not run on hardware
//
#include <hip/hip_runtime.h>

// ---------------------------------------------------------------------------
// Fused causal flash-attention for MI455X (gfx1250, wave32, WMMA).
//
// B=8, S=2048, D=1024, fp32 in/out.  scores = QK^T / 1024, causal, softmax, @V.
//
// Roofline: 137 GFLOP of matmul vs 256 MB of mandatory HBM traffic
// (11 us @ 23.3 TB/s) vs ~27 us at dense-f16 WMMA peak -> matrix-core bound.
// Strategy: flash-style fusion, fp32->f16 conversion once into LDS,
// v_wmma_f32_16x16x32_f16 everywhere, causal block skipping, K/V re-reads
// served by the 192 MB L2 (K+V total = 128 MB).
// ---------------------------------------------------------------------------

typedef _Float16 v16h __attribute__((ext_vector_type(16)));
typedef _Float16 v8h  __attribute__((ext_vector_type(8)));
typedef _Float16 v4h  __attribute__((ext_vector_type(4)));
typedef float    v8f  __attribute__((ext_vector_type(8)));

#define B_   8
#define S_   2048
#define D_   1024
#define BM   64          // query rows per workgroup
#define BN   32          // keys per inner block
#define NT_  256         // 8 wave32 waves

// LDS layout (bytes).  Strides padded for bank-conflict-free WMMA fragment
// loads (bank period = 256B; strides chosen coprime-ish with 64 banks) and
// 16B alignment for ds_load_b128.
#define QS_STRIDE 1032   // f16 elems per Q row   (2064 B, 16B aligned)
#define KS_STRIDE 1032   // f16 elems per K row
#define VS_STRIDE 40     // f16 elems per V^T row (80 B, 16B aligned)
#define SB_STRIDE 33     // f32 elems per S row
#define PB_STRIDE 40     // f16 elems per P row   (80 B, 16B aligned)

#define QS_OFF 0
#define KS_OFF (QS_OFF + BM * QS_STRIDE * 2)       // 132096
#define VS_OFF (KS_OFF + BN * KS_STRIDE * 2)       // 198144
#define SB_OFF (VS_OFF + D_ * VS_STRIDE * 2)       // 280064
#define PB_OFF (SB_OFF + BM * SB_STRIDE * 4)       // 288512
#define MB_OFF (PB_OFF + BM * PB_STRIDE * 2)       // 293632
#define LB_OFF (MB_OFF + BM * 4)
#define AB_OFF (LB_OFF + BM * 4)
#define SMEM_BYTES (AB_OFF + BM * 4)               // 294400 < 320 KB/WGP

__global__ __launch_bounds__(NT_, 1)
void fa_fwd_causal(const float* __restrict__ Qg, const float* __restrict__ Kg,
                   const float* __restrict__ Vg, float* __restrict__ Og)
{
    extern __shared__ char smem[];
    _Float16* Qs = (_Float16*)(smem + QS_OFF);
    _Float16* Ks = (_Float16*)(smem + KS_OFF);
    _Float16* Vs = (_Float16*)(smem + VS_OFF);   // transposed: [dim][key]
    float*    Sb = (float*)   (smem + SB_OFF);
    _Float16* Pb = (_Float16*)(smem + PB_OFF);
    float*    Mb = (float*)   (smem + MB_OFF);   // running row max
    float*    Lb = (float*)   (smem + LB_OFF);   // running row sum
    float*    Ab = (float*)   (smem + AB_OFF);   // per-block rescale alpha

    const int tid  = threadIdx.x;
    const int wave = tid >> 5;
    const int lane = tid & 31;
    const int hl   = lane >> 4;    // half-wave (selects WMMA K-chunk)
    const int l16  = lane & 15;

    const int m0    = blockIdx.x * BM;
    const int batch = blockIdx.y;

    const float* Qb = Qg + (size_t)batch * S_ * D_;
    const float* Kb = Kg + (size_t)batch * S_ * D_;
    const float* Vb = Vg + (size_t)batch * S_ * D_;
    float*       Ob = Og + (size_t)batch * S_ * D_;

    // ---- stage Q tile fp32 -> f16 into LDS (once, reused for all K blocks)
    #pragma unroll 4
    for (int it = 0; it < (BM * D_ / 4) / NT_; ++it) {   // 64 iters
        int g   = tid + it * NT_;
        int row = g >> 8;                 // 256 float4 groups per 1024-wide row
        int col = (g & 255) << 2;
        float4 v = *(const float4*)(Qb + (size_t)(m0 + row) * D_ + col);
        v4h h; h.x = (_Float16)v.x; h.y = (_Float16)v.y;
               h.z = (_Float16)v.z; h.w = (_Float16)v.w;
        *(v4h*)(Qs + row * QS_STRIDE + col) = h;
    }
    if (tid < BM) { Mb[tid] = -__builtin_inff(); Lb[tid] = 0.0f; Ab[tid] = 1.0f; }

    // O accumulator: each wave owns 4 M-tiles x 8 D-tiles (128-col slice)
    v8f acc[4][8] = {};

    __syncthreads();

    const int nb  = (m0 >> 5) + 2;    // causal: only key blocks <= diagonal
    const int miw = wave & 3;         // phase-1 tile coords for this wave
    const int niw = wave >> 2;

    for (int jb = 0; jb < nb; ++jb) {
        const int n0 = jb * BN;

        // ---- stage K block (row major f16)
        #pragma unroll 4
        for (int it = 0; it < (BN * D_ / 4) / NT_; ++it) {   // 32 iters
            int g = tid + it * NT_;
            int row = g >> 8; int col = (g & 255) << 2;
            float4 v = *(const float4*)(Kb + (size_t)(n0 + row) * D_ + col);
            v4h h; h.x = (_Float16)v.x; h.y = (_Float16)v.y;
                   h.z = (_Float16)v.z; h.w = (_Float16)v.w;
            *(v4h*)(Ks + row * KS_STRIDE + col) = h;
        }
        // ---- stage V block transposed ([dim][key]) so B-fragments are
        //      contiguous 32B ds_load_b128 pairs
        #pragma unroll 4
        for (int it = 0; it < (BN * D_ / 4) / NT_; ++it) {   // 32 iters
            int g = tid + it * NT_;
            int row = g >> 8; int col = (g & 255) << 2;
            float4 v = *(const float4*)(Vb + (size_t)(n0 + row) * D_ + col);
            Vs[(col + 0) * VS_STRIDE + row] = (_Float16)v.x;
            Vs[(col + 1) * VS_STRIDE + row] = (_Float16)v.y;
            Vs[(col + 2) * VS_STRIDE + row] = (_Float16)v.z;
            Vs[(col + 3) * VS_STRIDE + row] = (_Float16)v.w;
        }
        if (jb + 1 < nb) {   // warm L2/WGP$ for the next block
            __builtin_prefetch(Kb + (size_t)(n0 + BN) * D_ + tid * 128, 0, 1);
            __builtin_prefetch(Vb + (size_t)(n0 + BN) * D_ + tid * 128, 0, 1);
        }
        __syncthreads();

        // ---- phase 1: S-tile = Q K^T, one 16x16 tile per wave, 32 k-steps
        {
            v8f sacc = {};
            const _Float16* qrow = Qs + (miw * 16 + l16) * QS_STRIDE;
            const _Float16* krow = Ks + (niw * 16 + l16) * KS_STRIDE;
            #pragma unroll 8
            for (int kk = 0; kk < D_; kk += 32) {
                v16h a, b;
                v8h* ap = (v8h*)&a;
                ap[0] = *(const v8h*)(qrow + kk + hl * 8);        // K 0..7 / 8..15
                ap[1] = *(const v8h*)(qrow + kk + 16 + hl * 8);   // K 16..23 / 24..31
                v8h* bp = (v8h*)&b;
                bp[0] = *(const v8h*)(krow + kk + hl * 8);        // K 0..7 / 8..15
                bp[1] = *(const v8h*)(krow + kk + 16 + hl * 8);   // K 16..23 / 24..31
                sacc = __builtin_amdgcn_wmma_f32_16x16x32_f16(
                    false, a, false, b, (short)0, sacc, false, false);
            }
            #pragma unroll
            for (int v = 0; v < 8; ++v)
                Sb[(miw * 16 + hl * 8 + v) * SB_STRIDE + niw * 16 + l16] = sacc[v];
        }
        __syncthreads();

        // ---- phase 2: online softmax over the 64x32 tile (4 lanes / row)
        {
            const int r    = tid >> 2;
            const int seg  = tid & 3;
            const int rowg = m0 + r;
            const int c0   = seg << 3;
            float sv[8];
            float mx = -__builtin_inff();
            #pragma unroll
            for (int i = 0; i < 8; ++i) {
                float x = Sb[r * SB_STRIDE + c0 + i] * (1.0f / 1024.0f);
                x = (n0 + c0 + i <= rowg) ? x : -__builtin_inff();
                sv[i] = x;
                mx = fmaxf(mx, x);
            }
            mx = fmaxf(mx, __shfl_xor(mx, 1, 32));
            mx = fmaxf(mx, __shfl_xor(mx, 2, 32));
            float mold  = Mb[r];
            float mnew  = fmaxf(mold, mx);          // finite after first block
            float alpha = __expf(mold - mnew);      // exp(-inf)=0 on jb==0
            float sum = 0.0f;
            v8h ph;
            #pragma unroll
            for (int i = 0; i < 8; ++i) {
                float p = __expf(sv[i] - mnew);     // masked -> exp(-inf)=0
                sum += p;
                ph[i] = (_Float16)p;
            }
            *(v8h*)(Pb + r * PB_STRIDE + c0) = ph;  // 16B aligned store
            sum += __shfl_xor(sum, 1, 32);
            sum += __shfl_xor(sum, 2, 32);
            if (seg == 0) {
                Mb[r] = mnew;
                Lb[r] = Lb[r] * alpha + sum;
                Ab[r] = alpha;
            }
        }
        __syncthreads();

        // ---- phase 3: rescale O by alpha, then O += P x V (wave's 128-col slice)
        {
            #pragma unroll
            for (int mi = 0; mi < 4; ++mi) {
                float av[8];
                #pragma unroll
                for (int v = 0; v < 8; ++v) av[v] = Ab[mi * 16 + hl * 8 + v];
                #pragma unroll
                for (int nt = 0; nt < 8; ++nt)
                    #pragma unroll
                    for (int v = 0; v < 8; ++v)
                        acc[mi][nt][v] *= av[v];
            }
            v16h pa[4];
            #pragma unroll
            for (int mi = 0; mi < 4; ++mi) {
                const _Float16* prow = Pb + (mi * 16 + l16) * PB_STRIDE;
                v8h* ap = (v8h*)&pa[mi];
                ap[0] = *(const v8h*)(prow + hl * 8);
                ap[1] = *(const v8h*)(prow + 16 + hl * 8);
            }
            const int dbase = wave * 128;
            #pragma unroll
            for (int nt = 0; nt < 8; ++nt) {
                const _Float16* vrow = Vs + (dbase + nt * 16 + l16) * VS_STRIDE;
                v16h bv;
                v8h* bp = (v8h*)&bv;
                bp[0] = *(const v8h*)(vrow + hl * 8);
                bp[1] = *(const v8h*)(vrow + 16 + hl * 8);
                #pragma unroll
                for (int mi = 0; mi < 4; ++mi)
                    acc[mi][nt] = __builtin_amdgcn_wmma_f32_16x16x32_f16(
                        false, pa[mi], false, bv, (short)0, acc[mi][nt], false, false);
            }
        }
        __syncthreads();
    }

    // ---- epilogue: O /= l  and store fp32
    {
        const int dbase = wave * 128;
        #pragma unroll
        for (int mi = 0; mi < 4; ++mi) {
            float rl[8];
            #pragma unroll
            for (int v = 0; v < 8; ++v)
                rl[v] = 1.0f / Lb[mi * 16 + hl * 8 + v];
            #pragma unroll
            for (int nt = 0; nt < 8; ++nt) {
                #pragma unroll
                for (int v = 0; v < 8; ++v) {
                    int row = m0 + mi * 16 + hl * 8 + v;
                    int col = dbase + nt * 16 + l16;
                    Ob[(size_t)row * D_ + col] = acc[mi][nt][v] * rl[v];
                }
            }
        }
    }
}

extern "C" void kernel_launch(void* const* d_in, const int* in_sizes, int n_in,
                              void* d_out, int out_size, void* d_ws, size_t ws_size,
                              hipStream_t stream) {
    (void)in_sizes; (void)n_in; (void)d_ws; (void)ws_size; (void)out_size;
    const float* q = (const float*)d_in[0];
    const float* k = (const float*)d_in[1];
    const float* v = (const float*)d_in[2];
    float* out = (float*)d_out;

    // Allow large dynamic LDS (294 KB of the 320 KB / WGP). Idempotent host call.
    hipFuncSetAttribute((const void*)fa_fwd_causal,
                        hipFuncAttributeMaxDynamicSharedMemorySize, SMEM_BYTES);

    dim3 grid(S_ / BM, B_);     // 32 x 8 workgroups
    dim3 block(NT_);
    fa_fwd_causal<<<grid, block, SMEM_BYTES, stream>>>(q, k, v, out);
}
